// SPAModule_41051297415251
// MI455X (gfx1250) — matrix operand order repaired and checked by reading, not compiled, not run on hardware
//
#include <hip/hip_runtime.h>
#include <hip/hip_bf16.h>
#include <math.h>

// ---------------------------------------------------------------------------
// SPA module for MI455X (gfx1250): bf16 WMMA pipeline with async-LDS staging.
// B=4, N=1024, TN=3072, D=256, H=4, Dh=64, TOPK=64, TAU=1e-3
// ---------------------------------------------------------------------------

typedef __bf16 bf16;
typedef __attribute__((ext_vector_type(16))) __bf16 v16bf;
typedef __attribute__((ext_vector_type(2)))  __bf16 v2bf;
typedef __attribute__((ext_vector_type(8)))  float  v8f;

#define DIM   256
#define NTOK  1024
#define TN    3072
#define BATCH 4
#define ROWS  (BATCH * TN)      // 12288
#define TAU   1e-3f

// ---- WMMA fragment helpers (A/B in this file's contiguous k order, k = 16*(lane/16) + s) ------------------

__device__ __forceinline__ v8f wmma_bf16(v16bf a, v16bf b, v8f c) {
  // (neg_a, A, neg_b, B, c_mod, C, reuse_a, reuse_b)
  return __builtin_amdgcn_wmma_f32_16x16x32_bf16(false, a, false, b,
                                                 (short)0, c, false, false);
}

// A-matrix 16x32 (bf16), row-major source with leading dim `ld` (elements).
// Lane L (0..15): row M=L, K=0..15; lane L+16: row M=L, K=16..31.
__device__ __forceinline__ v16bf load_A(const bf16* __restrict__ p, int ld) {
  int lane = threadIdx.x & 31;
  int row  = lane & 15;
  int hi   = (lane >> 4) & 1;          // 0: lanes 0-15, 1: lanes 16-31
  const bf16* r = p + row * ld;
  v16bf a;
#pragma unroll
  for (int v = 0; v < 8; ++v) {
    v2bf lo  = *(const v2bf*)(r + hi * 16 + 2 * v);       // K = hi*16 + 2v
    a[2 * v]           = lo[0];
    a[2 * v + 1]       = lo[1];
  }
  return a;
}

// B-matrix 32x16 (bf16) where the K dimension is CONTIGUOUS in memory and the
// N (column) dimension has stride `ld`:  element(K,n) = p[n*ld + K].
// Lane L (0..15): col n=L, K=0..15; lane L+16: col n=L, K=16..31.
__device__ __forceinline__ v16bf load_BT(const bf16* __restrict__ p, int ld) {
  int lane = threadIdx.x & 31;
  int col  = lane & 15;
  int kb   = (lane >> 4) ? 16 : 0;
  const bf16* r = p + col * ld + kb;
  v16bf b;
#pragma unroll
  for (int v = 0; v < 8; ++v) {
    v2bf d = *(const v2bf*)(r + 2 * v);
    b[2 * v]     = d[0];
    b[2 * v + 1] = d[1];
  }
  return b;
}

// ---- gfx1250 async copy: one 16B chunk per lane, global -> LDS -------------
__device__ __forceinline__ void async_copy16(unsigned lds_off, const void* gaddr) {
  asm volatile("global_load_async_to_lds_b128 %0, %1, off"
               :: "v"(lds_off), "v"(gaddr) : "memory");
}

// ---- 1) per-(batch,channel) saliency: mean over tokens of |softthresh| -----
__global__ void sal_kernel(const float* __restrict__ Fc,
                           const float* __restrict__ Fs,
                           const float* __restrict__ Fa,
                           float* __restrict__ sal) {
  int b = blockIdx.x;
  int d = threadIdx.x;  // 256 threads
  float s = 0.f;
  for (int t = 0; t < NTOK; ++t) {
    int base = (b * NTOK + t) * DIM + d;
    float a0 = fabsf(Fc[base]) - TAU; s += a0 > 0.f ? a0 : 0.f;
    float a1 = fabsf(Fs[base]) - TAU; s += a1 > 0.f ? a1 : 0.f;
    float a2 = fabsf(Fa[base]) - TAU; s += a2 > 0.f ? a2 : 0.f;
  }
  sal[b * DIM + d] = s * (1.0f / (float)TN);
}

// ---- 2) top-64 channel mask (stable tie-break by lower index) --------------
__global__ void mask_kernel(const float* __restrict__ sal,
                            float* __restrict__ mask) {
  __shared__ float s[DIM];
  int b = blockIdx.x, d = threadIdx.x;
  s[d] = sal[b * DIM + d];
  __syncthreads();
  float v = s[d];
  int rank = 0;
  for (int j = 0; j < DIM; ++j) {
    float sj = s[j];
    if (sj > v || (sj == v && j < d)) rank++;
  }
  mask[b * DIM + d] = (rank < 64) ? 1.0f : 0.0f;
}

// ---- 3) soft-threshold + mask + cast to bf16 (concat layout (B,TN,D)) ------
__global__ void cast_kernel(const float* __restrict__ Fc,
                            const float* __restrict__ Fs,
                            const float* __restrict__ Fa,
                            const float* __restrict__ mask,
                            bf16* __restrict__ Xt) {
  int row = blockIdx.x;            // b*TN + t
  int d   = threadIdx.x;           // 256
  int b = row / TN, t = row % TN;
  const float* src = (t < NTOK) ? Fc : (t < 2 * NTOK ? Fs : Fa);
  int tl = t & (NTOK - 1);
  float x  = src[(b * NTOK + tl) * DIM + d];
  float ax = fabsf(x) - TAU;
  float xs = ax > 0.f ? (x > 0.f ? ax : -ax) : 0.f;
  xs *= mask[b * DIM + d];
  Xt[row * DIM + d] = (bf16)xs;
}

// ---- 4) weight f32 -> bf16 -------------------------------------------------
__global__ void castw_kernel(const float* __restrict__ W, bf16* __restrict__ Wb) {
  int i = blockIdx.x * blockDim.x + threadIdx.x;   // 65536 elems
  Wb[i] = (bf16)W[i];
}

// ---- 5) projection GEMM: Y = X @ W^T + bias, bf16 out ----------------------
// One wave (block of 32) per 16-row x 64-col output tile; K-chunk fragments
// are double-buffered in registers so loads overlap WMMA.
__global__ __launch_bounds__(32) void
proj_kernel(const bf16* __restrict__ X, const bf16* __restrict__ W,
            const float* __restrict__ bias, bf16* __restrict__ Y,
            int transposeV) {
  int bid   = blockIdx.x;           // 768 rowtiles * 4 ogroups = 3072
  int og    = bid & 3;
  int rt    = bid >> 2;
  int rbase = rt * 16;
  int obase = og * 64;
  int lane  = threadIdx.x;

  const bf16* Xr = X + rbase * DIM;
  const bf16* W0 = W + (obase + 0)  * DIM;
  const bf16* W1 = W + (obase + 16) * DIM;
  const bf16* W2 = W + (obase + 32) * DIM;
  const bf16* W3 = W + (obase + 48) * DIM;

  v8f acc[4] = {};
  v16bf a  = load_A(Xr, DIM);
  v16bf b0 = load_BT(W0, DIM);
  v16bf b1 = load_BT(W1, DIM);
  v16bf b2 = load_BT(W2, DIM);
  v16bf b3 = load_BT(W3, DIM);
#pragma unroll
  for (int kc = 0; kc < 8; ++kc) {
    int kn = (kc + 1 < 8) ? (kc + 1) * 32 : kc * 32;   // clamp: benign reload
    v16bf an  = load_A(Xr + kn, DIM);
    v16bf bn0 = load_BT(W0 + kn, DIM);
    v16bf bn1 = load_BT(W1 + kn, DIM);
    v16bf bn2 = load_BT(W2 + kn, DIM);
    v16bf bn3 = load_BT(W3 + kn, DIM);
    acc[0] = wmma_bf16(a, b0, acc[0]);
    acc[1] = wmma_bf16(a, b1, acc[1]);
    acc[2] = wmma_bf16(a, b2, acc[2]);
    acc[3] = wmma_bf16(a, b3, acc[3]);
    a = an; b0 = bn0; b1 = bn1; b2 = bn2; b3 = bn3;
  }

  int col   = lane & 15;
  int rhalf = (lane >> 4) ? 8 : 0;
#pragma unroll
  for (int f = 0; f < 4; ++f) {
    int   gcol = obase + f * 16 + col;
    float bv   = bias[gcol];
#pragma unroll
    for (int r = 0; r < 8; ++r) {
      float val  = acc[f][r] + bv;
      int   grow = rbase + rhalf + r;
      if (!transposeV) {
        Y[grow * DIM + gcol] = (bf16)val;
      } else {
        int b_ = grow / TN, t = grow % TN;
        Y[(b_ * DIM + gcol) * TN + t] = (bf16)val;   // (B, D, TN)
      }
    }
  }
}

// ---- 6) flash attention: one wave per (b, h, 16-query tile) ----------------
// K slabs (32 keys x 64 ch, 4KB) are double-buffered in LDS via
// global_load_async_to_lds_b128 / s_wait_asynccnt; V frags load early from
// global so they overlap softmax VALU work.
__global__ __launch_bounds__(32) void
attn_kernel(const bf16* __restrict__ Q, const bf16* __restrict__ K,
            const bf16* __restrict__ Vt, bf16* __restrict__ AO) {
  __shared__ bf16 kslab[2][32 * 64];   // 2 x 4KB async-staged K tiles
  __shared__ bf16 stage[16 * 32];      // P tile: C-layout -> A-layout

  int bid  = blockIdx.x;               // 4*4*192 = 3072
  int qt   = bid % 192;
  int h    = (bid / 192) & 3;
  int b    = bid / 768;
  int lane = threadIdx.x;
  int cb   = h * 64;                   // head channel base
  int qrow = b * TN + qt * 16;

  v16bf qa0 = load_A(Q + qrow * DIM + cb, DIM);
  v16bf qa1 = load_A(Q + qrow * DIM + cb + 32, DIM);

  v8f acc[4] = {};
  float m[8], l[8];
#pragma unroll
  for (int r = 0; r < 8; ++r) { m[r] = -3.0e38f; l[r] = 0.f; }

  int col   = lane & 15;
  int rhalf = (lane >> 4) ? 8 : 0;
  const float scale = 0.125f;          // 1/sqrt(64)

  const bf16* Khead  = K + ((size_t)b * TN) * DIM + cb;          // [key][ch]
  const bf16* Vhead  = Vt + ((size_t)b * DIM + cb) * TN;         // [ch][key]
  unsigned    kbase0 = (unsigned)(size_t)(void*)&kslab[0][0];
  unsigned    kbase1 = (unsigned)(size_t)(void*)&kslab[1][0];

  // issue async copy of one K slab: 32 rows x 128B, 8 x 16B chunks per lane
  auto issue_slab = [&](int kt, unsigned ldsb) {
    const char* gp = (const char*)(Khead + (size_t)kt * 32 * DIM);
#pragma unroll
    for (int i = 0; i < 8; ++i) {
      int chunk = lane + 32 * i;                 // 0..255
      int row   = chunk >> 3;
      int ro    = (chunk & 7) * 16;
      async_copy16(ldsb + chunk * 16, gp + (size_t)row * (DIM * 2) + ro);
    }
  };

  issue_slab(0, kbase0);

  for (int kt = 0; kt < 96; ++kt) {
    unsigned    curb = (kt & 1) ? kbase1 : kbase0;
    const bf16* kbuf = (kt & 1) ? &kslab[1][0] : &kslab[0][0];
    (void)curb;
    if (kt + 1 < 96) {
      issue_slab(kt + 1, (kt & 1) ? kbase0 : kbase1);
      asm volatile("s_wait_asynccnt 0x8" ::: "memory");  // slab kt complete
    } else {
      asm volatile("s_wait_asynccnt 0x0" ::: "memory");
    }

    // V fragments for this slab: load early so global latency overlaps VALU
    const bf16* vbase = Vhead + kt * 32;
    v16bf vf0 = load_BT(vbase + (size_t)0  * TN, TN);
    v16bf vf1 = load_BT(vbase + (size_t)16 * TN, TN);
    v16bf vf2 = load_BT(vbase + (size_t)32 * TN, TN);
    v16bf vf3 = load_BT(vbase + (size_t)48 * TN, TN);

    // logits: two 16x16 C tiles over the 32-key slab (K frags from LDS)
    v8f S0 = {}, S1 = {};
    {
      v16bf k00 = load_BT(kbuf + 0 * 64 + 0,  64);   // n0, c0
      v16bf k01 = load_BT(kbuf + 0 * 64 + 32, 64);   // n0, c1
      v16bf k10 = load_BT(kbuf + 16 * 64 + 0, 64);   // n1, c0
      v16bf k11 = load_BT(kbuf + 16 * 64 + 32, 64);  // n1, c1
      S0 = wmma_bf16(qa0, k00, S0);
      S0 = wmma_bf16(qa1, k01, S0);
      S1 = wmma_bf16(qa0, k10, S1);
      S1 = wmma_bf16(qa1, k11, S1);
    }

    // online softmax (row reductions across the 16-lane half owning each row)
    float corr[8];
#pragma unroll
    for (int r = 0; r < 8; ++r) {
      float s0 = S0[r] * scale, s1 = S1[r] * scale;
      float mx = fmaxf(s0, s1);
#pragma unroll
      for (int off = 1; off < 16; off <<= 1)
        mx = fmaxf(mx, __shfl_xor(mx, off, 32));
      float mn = fmaxf(m[r], mx);
      corr[r] = __expf(m[r] - mn);
      m[r] = mn;
      float p0 = __expf(s0 - mn), p1 = __expf(s1 - mn);
      float rs = p0 + p1;
#pragma unroll
      for (int off = 1; off < 16; off <<= 1)
        rs += __shfl_xor(rs, off, 32);
      l[r] = l[r] * corr[r] + rs;
      S0[r] = p0; S1[r] = p1;
    }
#pragma unroll
    for (int f = 0; f < 4; ++f)
#pragma unroll
      for (int r = 0; r < 8; ++r) acc[f][r] *= corr[r];

    // P: C-fragment layout -> LDS (row-major 16x32) -> A-fragment layout
#pragma unroll
    for (int r = 0; r < 8; ++r) {
      stage[(rhalf + r) * 32 + col]      = (bf16)S0[r];
      stage[(rhalf + r) * 32 + 16 + col] = (bf16)S1[r];
    }
    __syncthreads();                   // single-wave workgroup: orders LDS
    v16bf pfrag = load_A(stage, 32);
    __syncthreads();

    // P(16x32) @ V(32x64)
    acc[0] = wmma_bf16(pfrag, vf0, acc[0]);
    acc[1] = wmma_bf16(pfrag, vf1, acc[1]);
    acc[2] = wmma_bf16(pfrag, vf2, acc[2]);
    acc[3] = wmma_bf16(pfrag, vf3, acc[3]);
  }

  // normalize and store bf16 attention output, row-major (B*TN, D)
#pragma unroll
  for (int f = 0; f < 4; ++f)
#pragma unroll
    for (int r = 0; r < 8; ++r) {
      float val = acc[f][r] / l[r];
      AO[(qrow + rhalf + r) * DIM + cb + f * 16 + col] = (bf16)val;
    }
}

// ---- 7) output projection: f32, remapped to tuple-concatenated layout ------
__global__ __launch_bounds__(32) void
outproj_kernel(const bf16* __restrict__ AO, const bf16* __restrict__ W,
               const float* __restrict__ bias, float* __restrict__ out) {
  int bid   = blockIdx.x;           // 3072
  int og    = bid & 3;
  int rt    = bid >> 2;
  int rbase = rt * 16;
  int obase = og * 64;
  int lane  = threadIdx.x;

  const bf16* Ar = AO + rbase * DIM;
  const bf16* W0 = W + (obase + 0)  * DIM;
  const bf16* W1 = W + (obase + 16) * DIM;
  const bf16* W2 = W + (obase + 32) * DIM;
  const bf16* W3 = W + (obase + 48) * DIM;

  v8f acc[4] = {};
  v16bf a  = load_A(Ar, DIM);
  v16bf b0 = load_BT(W0, DIM);
  v16bf b1 = load_BT(W1, DIM);
  v16bf b2 = load_BT(W2, DIM);
  v16bf b3 = load_BT(W3, DIM);
#pragma unroll
  for (int kc = 0; kc < 8; ++kc) {
    int kn = (kc + 1 < 8) ? (kc + 1) * 32 : kc * 32;
    v16bf an  = load_A(Ar + kn, DIM);
    v16bf bn0 = load_BT(W0 + kn, DIM);
    v16bf bn1 = load_BT(W1 + kn, DIM);
    v16bf bn2 = load_BT(W2 + kn, DIM);
    v16bf bn3 = load_BT(W3 + kn, DIM);
    acc[0] = wmma_bf16(a, b0, acc[0]);
    acc[1] = wmma_bf16(a, b1, acc[1]);
    acc[2] = wmma_bf16(a, b2, acc[2]);
    acc[3] = wmma_bf16(a, b3, acc[3]);
    a = an; b0 = bn0; b1 = bn1; b2 = bn2; b3 = bn3;
  }

  int col   = lane & 15;
  int rhalf = (lane >> 4) ? 8 : 0;
#pragma unroll
  for (int f = 0; f < 4; ++f) {
    int   gcol = obase + f * 16 + col;
    float bv   = bias[gcol];
#pragma unroll
    for (int r = 0; r < 8; ++r) {
      int grow = rbase + rhalf + r;
      int b_ = grow / TN, t = grow % TN;
      int p  = t / NTOK, tl = t % NTOK;
      int orow = (p * BATCH + b_) * NTOK + tl;   // (out0|out1|out2) flat order
      out[orow * DIM + gcol] = acc[f][r] + bv;
    }
  }
}

// ---------------------------------------------------------------------------
extern "C" void kernel_launch(void* const* d_in, const int* in_sizes, int n_in,
                              void* d_out, int out_size, void* d_ws, size_t ws_size,
                              hipStream_t stream) {
  (void)in_sizes; (void)n_in; (void)out_size; (void)ws_size;

  const float* Fc = (const float*)d_in[0];
  const float* Fs = (const float*)d_in[1];
  const float* Fa = (const float*)d_in[2];
  const float* Wq = (const float*)d_in[3];
  const float* bq = (const float*)d_in[4];
  const float* Wk = (const float*)d_in[5];
  const float* bk = (const float*)d_in[6];
  const float* Wv = (const float*)d_in[7];
  const float* bv = (const float*)d_in[8];
  const float* Wo = (const float*)d_in[9];
  const float* bo = (const float*)d_in[10];
  float* out = (float*)d_out;

  // workspace layout
  char*  ws   = (char*)d_ws;
  float* sal  = (float*)ws;                          // 4 KB
  float* mask = (float*)(ws + 4096);                 // 4 KB
  bf16*  Wqb  = (bf16*)(ws + 8192);                  // 128 KB each
  bf16*  Wkb  = Wqb + DIM * DIM;
  bf16*  Wvb  = Wkb + DIM * DIM;
  bf16*  Wob  = Wvb + DIM * DIM;
  bf16*  Xt   = Wob + DIM * DIM;                     // (B*TN, D) bf16
  bf16*  Qb   = Xt + (size_t)ROWS * DIM;
  bf16*  Kb   = Qb + (size_t)ROWS * DIM;
  bf16*  Vtb  = Kb + (size_t)ROWS * DIM;             // (B, D, TN) bf16
  bf16*  AO   = Vtb + (size_t)ROWS * DIM;

  // 1-3: sparsify + select channels + cast
  sal_kernel<<<BATCH, DIM, 0, stream>>>(Fc, Fs, Fa, sal);
  mask_kernel<<<BATCH, DIM, 0, stream>>>(sal, mask);
  cast_kernel<<<ROWS, DIM, 0, stream>>>(Fc, Fs, Fa, mask, Xt);

  // 4: weights to bf16
  castw_kernel<<<DIM * DIM / 256, 256, 0, stream>>>(Wq, Wqb);
  castw_kernel<<<DIM * DIM / 256, 256, 0, stream>>>(Wk, Wkb);
  castw_kernel<<<DIM * DIM / 256, 256, 0, stream>>>(Wv, Wvb);
  castw_kernel<<<DIM * DIM / 256, 256, 0, stream>>>(Wo, Wob);

  // 5: Q/K/V projections (WMMA)
  proj_kernel<<<3072, 32, 0, stream>>>(Xt, Wqb, bq, Qb, 0);
  proj_kernel<<<3072, 32, 0, stream>>>(Xt, Wkb, bk, Kb, 0);
  proj_kernel<<<3072, 32, 0, stream>>>(Xt, Wvb, bv, Vtb, 1);

  // 6: fused multi-head attention (WMMA flash-style, async K staging)
  attn_kernel<<<3072, 32, 0, stream>>>(Qb, Kb, Vtb, AO);

  // 7: output projection + layout remap (WMMA)
  outproj_kernel<<<3072, 32, 0, stream>>>(AO, Wob, bo, out);
}
